// EncoderLayer_71141838291874
// MI455X (gfx1250) — compile-verified, matrix operand order checked by reading, not run on hardware
//
#include <hip/hip_runtime.h>

// ---------------------------------------------------------------------------
// CDNA5 (gfx1250, wave32) transformer encoder layer.
// GEMMs use v_wmma_f32_16x16x32_bf16; global->LDS staging is register
// double-buffered so next-tile HBM loads overlap ds_load + WMMA work.
// ---------------------------------------------------------------------------

typedef __attribute__((ext_vector_type(16))) __bf16 v16bf;
typedef __attribute__((ext_vector_type(8)))  __bf16 v8bf;
typedef __attribute__((ext_vector_type(8)))  float  v8f;

#define BM 128
#define BN 64
#define BK 32
#define BKP 40   // padded LDS stride in halves: 80 B (16B aligned, conflict-free)

#define B_   16
#define S_   512
#define D_   512
#define H_   8
#define F_   2048
#define DPH_ 64
#define SCALE_ 0.35355339059327373f   // (DPH//H)^-0.5 = 8^-0.5 (reference quirk)
#define LN_EPS_ 1e-5f

// Load one thread's share of the A tile (16 elems @ row arow, cols acb..acb+15)
// into two v8bf registers, converting fp32->bf16 if needed.
template<typename TA>
__device__ __forceinline__ void load_a_regs(const TA* __restrict__ A, int lda,
                                            int blockRow, int arow, int acb, int k0,
                                            v8bf& h0, v8bf& h1) {
  if constexpr (sizeof(TA) == 4) {
    const float* src = (const float*)A + (size_t)(blockRow + arow) * lda + k0 + acb;
    float4 f0 = *(const float4*)(src + 0);
    float4 f1 = *(const float4*)(src + 4);
    float4 f2 = *(const float4*)(src + 8);
    float4 f3 = *(const float4*)(src + 12);
    h0 = (v8bf){ (__bf16)f0.x,(__bf16)f0.y,(__bf16)f0.z,(__bf16)f0.w,
                 (__bf16)f1.x,(__bf16)f1.y,(__bf16)f1.z,(__bf16)f1.w };
    h1 = (v8bf){ (__bf16)f2.x,(__bf16)f2.y,(__bf16)f2.z,(__bf16)f2.w,
                 (__bf16)f3.x,(__bf16)f3.y,(__bf16)f3.z,(__bf16)f3.w };
  } else {
    const __bf16* src = (const __bf16*)A + (size_t)(blockRow + arow) * lda + k0 + acb;
    h0 = *(const v8bf*)(src + 0);
    h1 = *(const v8bf*)(src + 8);
  }
}

// ---------------------------------------------------------------------------
// Core WMMA tile GEMM: C[BMxBN] += A[BMxK] * Bt[BNxK]^T
// A row-major [M,K] (fp32 converted on stage, or bf16), Bt bf16 row-major [N,K].
// 256 threads = 8 waves; wave (wr,wc) in 4x2 grid owns 32x32 (2x2 fragments).
// ---------------------------------------------------------------------------
template<typename TA>
__device__ __forceinline__ void gemm_tile(const TA* __restrict__ A, int lda,
                                          const __bf16* __restrict__ Bt, int ldb,
                                          int K, int blockRow, int blockCol,
                                          v8f acc[2][2]) {
  __shared__ __bf16 As[BM][BKP];
  __shared__ __bf16 Bs[BN][BKP];

  const int tid  = threadIdx.x;
  const int lane = tid & 31;
  const int wave = tid >> 5;
  const int wr   = wave & 3;
  const int wc   = wave >> 2;
  const int lidx = lane & 15;
  const int ksel = (lane >> 4) * 8;   // ISA 16-bit A layout: lanes 16-31 hold K+8

  const int arow = tid >> 1;          // 0..127
  const int acb  = (tid & 1) * 16;    // 0 or 16
  const int brow = tid >> 2;          // 0..63
  const int bcb  = (tid & 3) * 8;     // 0,8,16,24

  // prime the register pipeline with tile k0 = 0
  v8bf ra0, ra1, rb0;
  load_a_regs<TA>(A, lda, blockRow, arow, acb, 0, ra0, ra1);
  rb0 = *(const v8bf*)(Bt + (size_t)(blockCol + brow) * ldb + bcb);

  for (int k0 = 0; k0 < K; k0 += BK) {
    // ---- commit current tile registers to LDS ----
    *(v8bf*)&As[arow][acb]     = ra0;
    *(v8bf*)&As[arow][acb + 8] = ra1;
    *(v8bf*)&Bs[brow][bcb]     = rb0;
    __syncthreads();

    // ---- issue next tile's global loads (overlap with ds_load + WMMA) ----
    if (k0 + BK < K) {
      load_a_regs<TA>(A, lda, blockRow, arow, acb, k0 + BK, ra0, ra1);
      rb0 = *(const v8bf*)(Bt + (size_t)(blockCol + brow) * ldb + (k0 + BK) + bcb);
    }
    if (k0 + 2 * BK < K) {  // speculative L2 prefetch two tiles ahead
      __builtin_prefetch((const char*)(A + (size_t)(blockRow + arow) * lda + k0 + 2 * BK + acb), 0, 0);
      __builtin_prefetch((const char*)(Bt + (size_t)(blockCol + brow) * ldb + k0 + 2 * BK + bcb), 0, 0);
    }

    // ---- fragments + WMMA ----
    v16bf a[2];
    v16bf b[2];
    #pragma unroll
    for (int fi = 0; fi < 2; ++fi) {
      const __bf16* p = &As[wr * 32 + fi * 16 + lidx][ksel];
      v8bf lo = *(const v8bf*)p;              // K = ksel .. ksel+7
      v8bf hi = *(const v8bf*)(p + 16);       // K = ksel+16 .. ksel+23
      a[fi] = __builtin_shufflevector(lo, hi, 0,1,2,3,4,5,6,7,8,9,10,11,12,13,14,15);
    }
    #pragma unroll
    for (int fj = 0; fj < 2; ++fj) {
      const __bf16* p = &Bs[wc * 32 + fj * 16 + lidx][ksel];
      v8bf lo = *(const v8bf*)p;
      v8bf hi = *(const v8bf*)(p + 16);
      b[fj] = __builtin_shufflevector(lo, hi, 0,1,2,3,4,5,6,7,8,9,10,11,12,13,14,15);
    }
    #pragma unroll
    for (int fi = 0; fi < 2; ++fi)
      #pragma unroll
      for (int fj = 0; fj < 2; ++fj)
        acc[fi][fj] = __builtin_amdgcn_wmma_f32_16x16x32_bf16(
            false, a[fi], false, b[fj], (short)0, acc[fi][fj], false, false);
    __syncthreads();
  }
}

// Iterate C fragment elements per the 16x16 f32 C layout (8 VGPRs x 32 lanes);
// invokes f(globalRow, globalCol, value) for each element this lane owns.
template<typename F>
__device__ __forceinline__ void frag_foreach(int blockRow, int blockCol,
                                             v8f acc[2][2], F&& f) {
  const int lane  = threadIdx.x & 31;
  const int wave  = threadIdx.x >> 5;
  const int wr    = wave & 3;
  const int wc    = wave >> 2;
  const int rbase = (lane >> 4) * 8;
  const int nidx  = lane & 15;
  #pragma unroll
  for (int fi = 0; fi < 2; ++fi)
    #pragma unroll
    for (int fj = 0; fj < 2; ++fj)
      #pragma unroll
      for (int i = 0; i < 8; ++i) {
        const int gr = blockRow + wr * 32 + fi * 16 + rbase + i;
        const int gc = blockCol + wc * 32 + fj * 16 + nidx;
        f(gr, gc, acc[fi][fj][i]);
      }
}

// ---------------------------------------------------------------------------
// Elementwise / reduction kernels
// ---------------------------------------------------------------------------

// Wt[n*K + k] = bf16(W[k*N + n])
__global__ void k_wtrans(const float* __restrict__ W, __bf16* __restrict__ Wt,
                         int K, int N) {
  int idx = blockIdx.x * blockDim.x + threadIdx.x;
  if (idx >= K * N) return;
  int k = idx / N;
  int n = idx % N;
  Wt[(size_t)n * K + k] = (__bf16)W[idx];
}

// t_bias[b,i,j] = sum_t cos((t_i - t_j)*w_t + b_t) * tw_t
__global__ void k_tbias(const float* __restrict__ tseq, const float* __restrict__ tw,
                        const float* __restrict__ tb, const float* __restrict__ twt,
                        float* __restrict__ tbias) {
  int idx = blockIdx.x * blockDim.x + threadIdx.x;   // < 16*512*512
  int b = idx >> 18;
  int i = (idx >> 9) & 511;
  int j = idx & 511;
  float dt = tseq[b * 512 + i] - tseq[b * 512 + j];
  float acc = 0.f;
  #pragma unroll
  for (int t = 0; t < 16; ++t)
    acc += cosf(dt * tw[t] + tb[t]) * twt[t];
  tbias[idx] = acc;
}

// softmax over rows of 512, one wave per row, in place
__global__ void k_softmax(float* __restrict__ attn) {
  int lane = threadIdx.x & 31;
  int wave = threadIdx.x >> 5;
  size_t row = (size_t)blockIdx.x * 8 + wave;
  float* p = attn + row * 512;
  float v[16];
  float mx = -1e30f;
  #pragma unroll
  for (int t = 0; t < 16; ++t) { v[t] = p[t * 32 + lane]; mx = fmaxf(mx, v[t]); }
  #pragma unroll
  for (int o = 16; o >= 1; o >>= 1) mx = fmaxf(mx, __shfl_xor(mx, o, 32));
  float sum = 0.f;
  #pragma unroll
  for (int t = 0; t < 16; ++t) { v[t] = expf(v[t] - mx); sum += v[t]; }
  #pragma unroll
  for (int o = 16; o >= 1; o >>= 1) sum += __shfl_xor(sum, o, 32);
  float inv = 1.0f / sum;
  #pragma unroll
  for (int t = 0; t < 16; ++t) p[t * 32 + lane] = v[t] * inv;
}

// out = LayerNorm(x + y) * g + b, one wave per 512-wide row
__global__ void k_add_ln(const float* __restrict__ x, const float* __restrict__ y,
                         const float* __restrict__ g, const float* __restrict__ bb,
                         float* __restrict__ out) {
  int lane = threadIdx.x & 31;
  int wave = threadIdx.x >> 5;
  size_t row = (size_t)blockIdx.x * 8 + wave;
  const float* px = x + row * 512;
  const float* py = y + row * 512;
  float v[16];
  float s = 0.f;
  float s2 = 0.f;
  #pragma unroll
  for (int t = 0; t < 16; ++t) {
    float u = px[t * 32 + lane] + py[t * 32 + lane];
    v[t] = u; s += u; s2 += u * u;
  }
  #pragma unroll
  for (int o = 16; o >= 1; o >>= 1) {
    s  += __shfl_xor(s,  o, 32);
    s2 += __shfl_xor(s2, o, 32);
  }
  float mean = s * (1.0f / 512.0f);
  float var  = s2 * (1.0f / 512.0f) - mean * mean;
  float inv  = rsqrtf(var + LN_EPS_);
  float* po = out + row * 512;
  #pragma unroll
  for (int t = 0; t < 16; ++t) {
    int c = t * 32 + lane;
    po[c] = (v[t] - mean) * inv * g[c] + bb[c];
  }
}

// ---------------------------------------------------------------------------
// GEMM kernels (epilogue variants)
// ---------------------------------------------------------------------------

// QKV projection; writes bf16 into the reference's raw-view head layout
// q[bh,s2,d2] with bh=b*8+(s>>6), s2=(s&63)*8+(c>>6), d2=c&63.
// WITH_VT additionally stores the transposed copy (for the context GEMM's B^T).
template<bool WITH_VT>
__global__ void k_gemm_qkv(const float* __restrict__ A, const __bf16* __restrict__ Wt,
                           const float* __restrict__ bias,
                           __bf16* __restrict__ outH, __bf16* __restrict__ outHT) {
  const int blockRow = blockIdx.y * BM;
  const int blockCol = blockIdx.x * BN;
  v8f acc[2][2] = {};
  gemm_tile<float>(A, D_, Wt, D_, D_, blockRow, blockCol, acc);
  frag_foreach(blockRow, blockCol, acc, [&](int gr, int gc, float val) {
    float v = val + bias[gc];
    int b  = gr >> 9;
    int s  = gr & 511;
    int jh = s >> 6;
    int srem = s & 63;
    int s2 = srem * 8 + (gc >> 6);
    int d2 = gc & 63;
    int bh = b * 8 + jh;
    outH[((size_t)(bh * 512 + s2)) * 64 + d2] = (__bf16)v;
    if constexpr (WITH_VT)
      outHT[((size_t)(bh * 64 + d2)) * 512 + s2] = (__bf16)v;  // v^T for context GEMM
  });
}

// scores = q @ k^T * SCALE + t_bias[bh % 16]  (per-bh via grid.z), fp32 out
__global__ void k_gemm_scores(const __bf16* __restrict__ q, const __bf16* __restrict__ kk,
                              const float* __restrict__ tbias, float* __restrict__ attn) {
  const int z = blockIdx.z;
  const int blockRow = blockIdx.y * BM;
  const int blockCol = blockIdx.x * BN;
  v8f acc[2][2] = {};
  gemm_tile<__bf16>(q + (size_t)z * S_ * DPH_, DPH_,
                    kk + (size_t)z * S_ * DPH_, DPH_,
                    DPH_, blockRow, blockCol, acc);
  const float* tb = tbias + (size_t)(z & 15) * S_ * S_;
  float* out = attn + (size_t)z * S_ * S_;
  frag_foreach(blockRow, blockCol, acc, [&](int gr, int gc, float val) {
    out[(size_t)gr * S_ + gc] = val * SCALE_ + tb[(size_t)gr * S_ + gc];
  });
}

// context = attn @ v  (A fp32 converted on stage; Bt = v^T bf16), per-bh
__global__ void k_gemm_context(const float* __restrict__ attn, const __bf16* __restrict__ vt,
                               float* __restrict__ ctx) {
  const int z = blockIdx.z;
  const int blockRow = blockIdx.y * BM;
  const int blockCol = 0;
  v8f acc[2][2] = {};
  gemm_tile<float>(attn + (size_t)z * S_ * S_, S_,
                   vt + (size_t)z * DPH_ * S_, S_,
                   S_, blockRow, blockCol, acc);
  float* out = ctx + (size_t)z * S_ * DPH_;   // raw-view back to [B,S,D]
  frag_foreach(blockRow, blockCol, acc, [&](int gr, int gc, float val) {
    out[(size_t)gr * DPH_ + gc] = val;
  });
}

// C = A @ W + bias, fp32 A, fp32 out
__global__ void k_gemm_bias(const float* __restrict__ A, const __bf16* __restrict__ Wt,
                            const float* __restrict__ bias, float* __restrict__ C,
                            int K, int N) {
  const int blockRow = blockIdx.y * BM;
  const int blockCol = blockIdx.x * BN;
  v8f acc[2][2] = {};
  gemm_tile<float>(A, K, Wt, K, K, blockRow, blockCol, acc);
  frag_foreach(blockRow, blockCol, acc, [&](int gr, int gc, float val) {
    C[(size_t)gr * N + gc] = val + bias[gc];
  });
}

// C = A @ W + bias, bf16 A, fp32 out (FFN second matmul)
__global__ void k_gemm_bias_bf16A(const __bf16* __restrict__ A, const __bf16* __restrict__ Wt,
                                  const float* __restrict__ bias, float* __restrict__ C,
                                  int K, int N) {
  const int blockRow = blockIdx.y * BM;
  const int blockCol = blockIdx.x * BN;
  v8f acc[2][2] = {};
  gemm_tile<__bf16>(A, K, Wt, K, K, blockRow, blockCol, acc);
  frag_foreach(blockRow, blockCol, acc, [&](int gr, int gc, float val) {
    C[(size_t)gr * N + gc] = val + bias[gc];
  });
}

// C = gelu(A @ W + bias) -> bf16 (FFN first matmul)
__global__ void k_gemm_gelu(const float* __restrict__ A, const __bf16* __restrict__ Wt,
                            const float* __restrict__ bias, __bf16* __restrict__ C,
                            int K, int N) {
  const int blockRow = blockIdx.y * BM;
  const int blockCol = blockIdx.x * BN;
  v8f acc[2][2] = {};
  gemm_tile<float>(A, K, Wt, K, K, blockRow, blockCol, acc);
  frag_foreach(blockRow, blockCol, acc, [&](int gr, int gc, float val) {
    float x = val + bias[gc];
    float g = 0.5f * x * (1.0f + tanhf(0.7978845608028654f * (x + 0.044715f * x * x * x)));
    C[(size_t)gr * N + gc] = (__bf16)g;
  });
}

// ---------------------------------------------------------------------------
// Host launch
// ---------------------------------------------------------------------------
extern "C" void kernel_launch(void* const* d_in, const int* in_sizes, int n_in,
                              void* d_out, int out_size, void* d_ws, size_t ws_size,
                              hipStream_t stream) {
  (void)in_sizes; (void)n_in; (void)out_size; (void)ws_size;

  const float* inputs  = (const float*)d_in[0];
  const float* t_seq   = (const float*)d_in[1];
  const float* Wq      = (const float*)d_in[2];
  const float* bq      = (const float*)d_in[3];
  const float* Wk      = (const float*)d_in[4];
  const float* bk      = (const float*)d_in[5];
  const float* Wv      = (const float*)d_in[6];
  const float* bv      = (const float*)d_in[7];
  const float* Wo      = (const float*)d_in[8];
  const float* bo      = (const float*)d_in[9];
  const float* ln1_g   = (const float*)d_in[10];
  const float* ln1_b   = (const float*)d_in[11];
  const float* time_w  = (const float*)d_in[12];
  const float* time_b  = (const float*)d_in[13];
  const float* time_wt = (const float*)d_in[14];
  const float* W1      = (const float*)d_in[15];
  const float* b1f     = (const float*)d_in[16];
  const float* W2      = (const float*)d_in[17];
  const float* b2f     = (const float*)d_in[18];
  const float* ln2_g   = (const float*)d_in[19];
  const float* ln2_b   = (const float*)d_in[20];

  const size_t M = (size_t)B_ * S_;            // 8192
  float* out2 = (float*)d_out;                 // [8192*512]
  float* attn = (float*)d_out + M * D_;        // [128*512*512]

  char* ws = (char*)d_ws;
  size_t off = 0;
  auto alloc = [&](size_t bytes) -> void* {
    void* p = ws + off;
    off += (bytes + 255) & ~(size_t)255;
    return p;
  };
  __bf16* WqT   = (__bf16*)alloc((size_t)D_ * D_ * 2);
  __bf16* WkT   = (__bf16*)alloc((size_t)D_ * D_ * 2);
  __bf16* WvT   = (__bf16*)alloc((size_t)D_ * D_ * 2);
  __bf16* WoT   = (__bf16*)alloc((size_t)D_ * D_ * 2);
  __bf16* W1T   = (__bf16*)alloc((size_t)D_ * F_ * 2);
  __bf16* W2T   = (__bf16*)alloc((size_t)F_ * D_ * 2);
  __bf16* qh    = (__bf16*)alloc(M * D_ * 2);
  __bf16* kh    = (__bf16*)alloc(M * D_ * 2);
  __bf16* vh    = (__bf16*)alloc(M * D_ * 2);
  __bf16* vth   = (__bf16*)alloc(M * D_ * 2);
  float*  tbias = (float*) alloc((size_t)B_ * S_ * S_ * 4);
  float*  ctx   = (float*) alloc(M * D_ * 4);
  float*  aout  = (float*) alloc(M * D_ * 4);
  float*  out1  = (float*) alloc(M * D_ * 4);
  __bf16* ff    = (__bf16*)alloc(M * F_ * 2);
  float*  ff2   = (float*) alloc(M * D_ * 4);

  dim3 blk(256);

  // 1) weight transpose + bf16 convert
  k_wtrans<<<dim3((D_ * D_) / 256), blk, 0, stream>>>(Wq, WqT, D_, D_);
  k_wtrans<<<dim3((D_ * D_) / 256), blk, 0, stream>>>(Wk, WkT, D_, D_);
  k_wtrans<<<dim3((D_ * D_) / 256), blk, 0, stream>>>(Wv, WvT, D_, D_);
  k_wtrans<<<dim3((D_ * D_) / 256), blk, 0, stream>>>(Wo, WoT, D_, D_);
  k_wtrans<<<dim3((D_ * F_) / 256), blk, 0, stream>>>(W1, W1T, D_, F_);
  k_wtrans<<<dim3((F_ * D_) / 256), blk, 0, stream>>>(W2, W2T, F_, D_);

  // 2) additive time bias [B,S,S]
  k_tbias<<<dim3((B_ * S_ * S_) / 256), blk, 0, stream>>>(t_seq, time_w, time_b, time_wt, tbias);

  // 3) QKV projections (bf16 WMMA), head-split bf16 outputs (+ v^T)
  k_gemm_qkv<false><<<dim3(D_ / BN, M / BM), blk, 0, stream>>>(inputs, WqT, bq, qh, nullptr);
  k_gemm_qkv<false><<<dim3(D_ / BN, M / BM), blk, 0, stream>>>(inputs, WkT, bk, kh, nullptr);
  k_gemm_qkv<true ><<<dim3(D_ / BN, M / BM), blk, 0, stream>>>(inputs, WvT, bv, vh, vth);

  // 4) attention scores -> d_out attn region (fused scale + time bias)
  k_gemm_scores<<<dim3(S_ / BN, S_ / BM, B_ * H_), blk, 0, stream>>>(qh, kh, tbias, attn);

  // 5) softmax in place (attn is also a returned output)
  k_softmax<<<dim3((B_ * H_ * S_) / 8), blk, 0, stream>>>(attn);

  // 6) context = attn @ v
  k_gemm_context<<<dim3(1, S_ / BM, B_ * H_), blk, 0, stream>>>(attn, vth, ctx);

  // 7) output projection
  k_gemm_bias<<<dim3(D_ / BN, M / BM), blk, 0, stream>>>(ctx, WoT, bo, aout, D_, D_);

  // 8) LN1(residual + attn_out)
  k_add_ln<<<dim3(M / 8), blk, 0, stream>>>(inputs, aout, ln1_g, ln1_b, out1);

  // 9) FFN: gelu(out1 @ W1 + b1) -> bf16
  k_gemm_gelu<<<dim3(F_ / BN, M / BM), blk, 0, stream>>>(out1, W1T, b1f, ff, D_, F_);

  // 10) FFN second matmul (bf16 A)
  k_gemm_bias_bf16A<<<dim3(D_ / BN, M / BM), blk, 0, stream>>>(ff, W2T, b2f, ff2, F_, D_);

  // 11) LN2(out1 + ff) -> out2 (first chunk of d_out)
  k_add_ln<<<dim3(M / 8), blk, 0, stream>>>(out1, ff2, ln2_g, ln2_b, out2);
}
